// EBConv2d_62234076119374
// MI455X (gfx1250) — matrix operand order repaired and checked by reading, not compiled, not run on hardware
//
#include <hip/hip_runtime.h>
#include <math.h>

typedef __attribute__((ext_vector_type(16))) _Float16 v16h;
typedef __attribute__((ext_vector_type(8)))  float    v8f;

#define BSZ   32
#define CCH   128
#define OCH   128
#define HH    112
#define WW    112
#define HWSZ  (HH*WW)        // 12544
#define KTOT  (CCH*9)        // 1152
#define KDWT  (KTOT/2)       // 576 dword-pairs per weight row
#define KSTEP 32
#define NKT   (KTOT/KSTEP)   // 36
#define NT2   (HWSZ/128)     // 98 exact N tiles of 128

union H2    { _Float16 h[2]; unsigned int u; };
union FragU { unsigned int u[8]; v16h v; };

// split fp32 pair -> packed-f16 hi dword + lo dword (x ~= hi + lo); used by aprep and fallback conv
__device__ __forceinline__ unsigned int splitPack(float a, float b, unsigned int& lopack) {
    _Float16 ha = (_Float16)a;
    _Float16 hb = (_Float16)b;
    _Float16 la = (_Float16)(a - (float)ha);
    _Float16 lb = (_Float16)(b - (float)hb);
    H2 hi; hi.h[0] = ha; hi.h[1] = hb;
    H2 lo; lo.h[0] = la; lo.h[1] = lb;
    lopack = lo.u;
    return hi.u;
}
// single element -> dword {f16 hi, f16 lo}
__device__ __forceinline__ unsigned int splitOne(float a) {
    _Float16 ha = (_Float16)a;
    _Float16 la = (_Float16)(a - (float)ha);
    H2 p; p.h[0] = ha; p.h[1] = la;
    return p.u;
}

// ------------- kernel 1a: avg pool + produce packed split image -------------
__global__ __launch_bounds__(256) void avgsplit_kernel(const float* __restrict__ x,
                                                       float* __restrict__ avg,
                                                       unsigned int* __restrict__ xs) {
    __shared__ float red[256];
    int bc = blockIdx.x;
    const float* p = x + (size_t)bc * HWSZ;
    unsigned int* q = xs + (size_t)bc * HWSZ;
    float s = 0.f;
    for (int i = threadIdx.x; i < HWSZ; i += 256) {
        float v = p[i];
        s += v;
        q[i] = splitOne(v);
    }
    red[threadIdx.x] = s;
    __syncthreads();
    for (int off = 128; off > 0; off >>= 1) {
        if ((int)threadIdx.x < off) red[threadIdx.x] += red[threadIdx.x + off];
        __syncthreads();
    }
    if (threadIdx.x == 0) avg[bc] = red[0] * (1.0f / (float)HWSZ);
}

// ------------- kernel 1b (fallback): avg pool only -------------
__global__ __launch_bounds__(256) void avg_kernel(const float* __restrict__ x,
                                                  float* __restrict__ avg) {
    __shared__ float red[256];
    int bc = blockIdx.x;
    const float* p = x + (size_t)bc * HWSZ;
    float s = 0.f;
    for (int i = threadIdx.x; i < HWSZ; i += 256) s += p[i];
    red[threadIdx.x] = s;
    __syncthreads();
    for (int off = 128; off > 0; off >>= 1) {
        if ((int)threadIdx.x < off) red[threadIdx.x] += red[threadIdx.x + off];
        __syncthreads();
    }
    if (threadIdx.x == 0) avg[bc] = red[0] * (1.0f / (float)HWSZ);
}

// ------------- kernel 1c: prepack weight[0] -> hi/lo f16 planes -------------
__global__ __launch_bounds__(256) void aprep_kernel(const float* __restrict__ wgt,
                                                    unsigned int* __restrict__ Ahp,
                                                    unsigned int* __restrict__ Alp) {
    int q = blockIdx.x * 256 + threadIdx.x;       // 128*576 tasks
    const float2 v2 = *(const float2*)(wgt + (size_t)q * 2);
    unsigned int lo, hi = splitPack(v2.x, v2.y, lo);
    Ahp[q] = hi;  Alp[q] = lo;
}

// ------------- kernel 2: SE MLP -------------
__global__ __launch_bounds__(128) void se_kernel(const float* __restrict__ avg,
                                                 const float* __restrict__ w1,
                                                 const float* __restrict__ w2,
                                                 float* __restrict__ se) {
    __shared__ float a[CCH];
    __shared__ float hdn[16];
    int b = blockIdx.x, t = threadIdx.x;
    a[t] = avg[b * CCH + t];
    __syncthreads();
    if (t < 16) {
        float s = 0.f;
        for (int c = 0; c < CCH; ++c) s += a[c] * w1[t * CCH + c];
        hdn[t] = s > 0.f ? s : 0.f;
    }
    __syncthreads();
    float s = 0.f;
    #pragma unroll
    for (int j = 0; j < 16; ++j) s += hdn[j] * w2[t * 16 + j];
    se[b * OCH + t] = 1.0f / (1.0f + expf(-s));
}

// ------------- kernel 3: 128x128-tile implicit-GEMM conv, prepacked split-f16 -------------
__global__ __launch_bounds__(256) void conv_wmma_kernel(
    const unsigned int* __restrict__ xs,    // packed {hi,lo} image [B,C,H,W]
    const unsigned int* __restrict__ Ahp,   // weight hi plane [O][576]
    const unsigned int* __restrict__ Alp,   // weight lo plane [O][576]
    const float* __restrict__ bias,
    const float* __restrict__ se,
    float* __restrict__ out)
{
    __shared__ unsigned int Ahi[2][128][16], Alo[2][128][16];   // 32KB
    __shared__ unsigned int Bhi[2][128][16], Blo[2][128][16];   // 32KB

    int idx = blockIdx.x;
    int nt  = idx % NT2;
    int b   = idx / NT2;

    int tid   = threadIdx.x;
    int lane  = tid & 31;
    int wave  = tid >> 5;
    int waveM = wave >> 1;     // 0..3 -> 32 M rows
    int waveN = wave & 1;      // 0..1 -> 64 N cols
    int half  = lane >> 4;
    int l16   = lane & 15;

    int Nbase = nt * 128;
    const unsigned int* xb = xs + (size_t)b * CCH * HWSZ;   // uniform per block

    // ---- loader tasks: A 2048 dwords/plane, B 2048 dword-pairs -> 8 each ----
    int mA[8], kdA[8];  unsigned int aoffG[8];
    #pragma unroll
    for (int i = 0; i < 8; ++i) {
        int q = tid + 256 * i;
        mA[i] = q >> 4;  kdA[i] = q & 15;
        aoffG[i] = (unsigned int)mA[i] * KDWT + kdA[i];    // dword index; +16/step
    }
    int nB[8], kdB[8], cB[8], remB[8];
    unsigned int obase[8], m9[8];
    #pragma unroll
    for (int i = 0; i < 8; ++i) {
        int q = tid + 256 * i;
        nB[i] = q >> 4;  kdB[i] = q & 15;
        int p  = Nbase + nB[i];
        int oh = p / WW, ow = p % WW;
        obase[i] = (unsigned int)(oh * WW + ow);
        int k = kdB[i] * 2;
        cB[i] = k / 9;  remB[i] = k % 9;
        unsigned int rowm = (ow > 0 ? 1u : 0u) | 2u | (ow < WW - 1 ? 4u : 0u);
        m9[i] = (oh > 0 ? rowm : 0u) | (rowm << 3) | (oh < HH - 1 ? (rowm << 6) : 0u);
    }

    v8f acc[2][4];
    #pragma unroll
    for (int mi = 0; mi < 2; ++mi)
        #pragma unroll
        for (int ni = 0; ni < 4; ++ni) acc[mi][ni] = (v8f){};

    unsigned int aSh[8], aSl[8];        // staged A dwords (hi/lo planes)
    unsigned int bS0[8], bS1[8];        // staged B dwords (elements k, k+1)

    #define ISSUE_TILE()                                                           \
    {                                                                              \
        _Pragma("unroll")                                                          \
        for (int i = 0; i < 8; ++i) {                                              \
            aSh[i] = Ahp[aoffG[i]];  aSl[i] = Alp[aoffG[i]];                       \
            aoffG[i] += 16;                                                        \
        }                                                                          \
        _Pragma("unroll")                                                          \
        for (int i = 0; i < 8; ++i) {                                              \
            int c = cB[i], rem = remB[i];                                          \
            unsigned int d0 = 0u, d1 = 0u;                                         \
            {                                                                      \
                int kh = rem / 3, kw = rem - kh * 3;                               \
                if ((m9[i] >> rem) & 1)                                            \
                    d0 = xb[(int)obase[i] + c * HWSZ + (kh - 1) * WW + (kw - 1)];  \
            }                                                                      \
            int rem1 = rem + 1, c1 = c;                                            \
            if (rem1 == 9) { rem1 = 0; c1 = c + 1; }                               \
            {                                                                      \
                int kh = rem1 / 3, kw = rem1 - kh * 3;                             \
                if ((m9[i] >> rem1) & 1)                                           \
                    d1 = xb[(int)obase[i] + c1 * HWSZ + (kh - 1) * WW + (kw - 1)]; \
            }                                                                      \
            bS0[i] = d0;  bS1[i] = d1;                                             \
            rem += 5; c += 3;                                                      \
            if (rem >= 9) { rem -= 9; ++c; }                                       \
            cB[i] = c;  remB[i] = rem;                                             \
        }                                                                          \
    }

    ISSUE_TILE();

    for (int kt = 0; kt < NKT; ++kt) {
        int buf = kt & 1;

        // ---- stage regs -> LDS[buf] (A: pure moves; B: 2x v_perm repack) ----
        #pragma unroll
        for (int i = 0; i < 8; ++i) {
            Ahi[buf][mA[i]][kdA[i]] = aSh[i];
            Alo[buf][mA[i]][kdA[i]] = aSl[i];
        }
        #pragma unroll
        for (int i = 0; i < 8; ++i) {
            // bS*: {f16 hi, f16 lo}; repack into {hi(k),hi(k+1)} / {lo(k),lo(k+1)}
            unsigned int hp = __builtin_amdgcn_perm(bS1[i], bS0[i], 0x05040100u);
            unsigned int lp = __builtin_amdgcn_perm(bS1[i], bS0[i], 0x07060302u);
            Bhi[buf][nB[i]][kdB[i]] = hp;
            Blo[buf][nB[i]][kdB[i]] = lp;
        }
        __syncthreads();            // drains DScnt -> 2-buffer rotation race-free

        if (kt + 1 < NKT) ISSUE_TILE();   // next tile's global loads in flight

        // ---- A fragments (16-bit WMMA layout: Kdw {aoff..+3, aoff+8..+11}) ----
        int arow0 = waveM * 32 + l16, arow1 = arow0 + 16;
        int aoff  = half * 4;
        FragU a0h, a0l, a1h, a1l;
        *(uint4*)&a0h.u[0] = *(const uint4*)&Ahi[buf][arow0][aoff];
        *(uint4*)&a0h.u[4] = *(const uint4*)&Ahi[buf][arow0][aoff + 8];
        *(uint4*)&a0l.u[0] = *(const uint4*)&Alo[buf][arow0][aoff];
        *(uint4*)&a0l.u[4] = *(const uint4*)&Alo[buf][arow0][aoff + 8];
        *(uint4*)&a1h.u[0] = *(const uint4*)&Ahi[buf][arow1][aoff];
        *(uint4*)&a1h.u[4] = *(const uint4*)&Ahi[buf][arow1][aoff + 8];
        *(uint4*)&a1l.u[0] = *(const uint4*)&Alo[buf][arow1][aoff];
        *(uint4*)&a1l.u[4] = *(const uint4*)&Alo[buf][arow1][aoff + 8];

        int boff = half * 4;
        #pragma unroll
        for (int ni = 0; ni < 4; ++ni) {
            int brow = waveN * 64 + ni * 16 + l16;
            FragU bh, bl;
            *(uint4*)&bh.u[0] = *(const uint4*)&Bhi[buf][brow][boff];
            *(uint4*)&bh.u[4] = *(const uint4*)&Bhi[buf][brow][boff + 8];
            *(uint4*)&bl.u[0] = *(const uint4*)&Blo[buf][brow][boff];
            *(uint4*)&bl.u[4] = *(const uint4*)&Blo[buf][brow][boff + 8];

            acc[0][ni] = __builtin_amdgcn_wmma_f32_16x16x32_f16(false, a0h.v, false, bh.v, (short)0, acc[0][ni], false, false);
            acc[0][ni] = __builtin_amdgcn_wmma_f32_16x16x32_f16(false, a0h.v, false, bl.v, (short)0, acc[0][ni], false, false);
            acc[0][ni] = __builtin_amdgcn_wmma_f32_16x16x32_f16(false, a0l.v, false, bh.v, (short)0, acc[0][ni], false, false);
            acc[1][ni] = __builtin_amdgcn_wmma_f32_16x16x32_f16(false, a1h.v, false, bh.v, (short)0, acc[1][ni], false, false);
            acc[1][ni] = __builtin_amdgcn_wmma_f32_16x16x32_f16(false, a1h.v, false, bl.v, (short)0, acc[1][ni], false, false);
            acc[1][ni] = __builtin_amdgcn_wmma_f32_16x16x32_f16(false, a1l.v, false, bh.v, (short)0, acc[1][ni], false, false);
        }
    }

    // ---- epilogue ----
    #pragma unroll
    for (int mi = 0; mi < 2; ++mi) {
        #pragma unroll
        for (int r = 0; r < 8; ++r) {
            int o = waveM * 32 + mi * 16 + half * 8 + r;
            float sc = se[b * OCH + o];
            float bb = bias[o];
            size_t rb = (size_t)(b * OCH + o) * HWSZ;
            #pragma unroll
            for (int ni = 0; ni < 4; ++ni) {
                int n = Nbase + waveN * 64 + ni * 16 + l16;
                out[rb + n] = (acc[mi][ni][r] + bb) * sc;
            }
        }
    }
    #undef ISSUE_TILE
}

// ------------- fallback conv (in-kernel split, 64x128 tiles) for small ws -------------
__global__ __launch_bounds__(256) void conv_wmma_fb(
    const float* __restrict__ x, const float* __restrict__ wgt,
    const float* __restrict__ bias, const float* __restrict__ se,
    float* __restrict__ out)
{
    __shared__ unsigned int Ahi[2][64][16],  Alo[2][64][16];
    __shared__ unsigned int Bhi[2][128][16], Blo[2][128][16];

    int idx = blockIdx.x;
    int nt  = idx % NT2;
    int mt  = (idx / NT2) & 1;
    int b   = idx / (2 * NT2);
    int tid = threadIdx.x, lane = tid & 31, wave = tid >> 5;
    int waveM = wave >> 2, waveN = wave & 3, half = lane >> 4, l16 = lane & 15;
    int Mbase = mt * 64, Nbase = nt * 128;
    const float* xb = x + (size_t)b * CCH * HWSZ;

    int mA[4], kdA[4]; unsigned int aoffG[4];
    #pragma unroll
    for (int i = 0; i < 4; ++i) {
        int q = tid + 256 * i;
        mA[i] = q >> 4; kdA[i] = q & 15;
        aoffG[i] = (unsigned int)(Mbase + mA[i]) * KTOT + kdA[i] * 2;
    }
    int nB[8], kdB[8], cB[8], remB[8];
    unsigned int obase[8], m9[8];
    #pragma unroll
    for (int i = 0; i < 8; ++i) {
        int q = tid + 256 * i;
        nB[i] = q >> 4; kdB[i] = q & 15;
        int p = Nbase + nB[i];
        int oh = p / WW, ow = p % WW;
        obase[i] = (unsigned int)(oh * WW + ow);
        int k = kdB[i] * 2;
        cB[i] = k / 9; remB[i] = k % 9;
        unsigned int rowm = (ow > 0 ? 1u : 0u) | 2u | (ow < WW - 1 ? 4u : 0u);
        m9[i] = (oh > 0 ? rowm : 0u) | (rowm << 3) | (oh < HH - 1 ? (rowm << 6) : 0u);
    }
    v8f acc00 = {}, acc01 = {}, acc10 = {}, acc11 = {};
    float aS[4][2], bS[8][2];

    #define ISSUE_FB()                                                            \
    {                                                                             \
        _Pragma("unroll")                                                         \
        for (int i = 0; i < 4; ++i) {                                             \
            const float2 v2 = *(const float2*)(wgt + aoffG[i]);                   \
            aS[i][0] = v2.x; aS[i][1] = v2.y;  aoffG[i] += KSTEP;                 \
        }                                                                         \
        _Pragma("unroll")                                                         \
        for (int i = 0; i < 8; ++i) {                                             \
            int c = cB[i], rem = remB[i];                                         \
            float v0 = 0.f, v1 = 0.f;                                             \
            { int kh = rem / 3, kw = rem - kh * 3;                                \
              if ((m9[i] >> rem) & 1)                                             \
                  v0 = xb[(int)obase[i] + c * HWSZ + (kh - 1) * WW + (kw - 1)]; } \
            int rem1 = rem + 1, c1 = c;                                           \
            if (rem1 == 9) { rem1 = 0; c1 = c + 1; }                              \
            { int kh = rem1 / 3, kw = rem1 - kh * 3;                              \
              if ((m9[i] >> rem1) & 1)                                            \
                  v1 = xb[(int)obase[i] + c1 * HWSZ + (kh - 1) * WW + (kw - 1)]; }\
            bS[i][0] = v0; bS[i][1] = v1;                                         \
            rem += 5; c += 3;                                                     \
            if (rem >= 9) { rem -= 9; ++c; }                                      \
            cB[i] = c; remB[i] = rem;                                             \
        }                                                                         \
    }

    ISSUE_FB();
    for (int kt = 0; kt < NKT; ++kt) {
        int buf = kt & 1;
        #pragma unroll
        for (int i = 0; i < 4; ++i) {
            unsigned int lo, hi = splitPack(aS[i][0], aS[i][1], lo);
            Ahi[buf][mA[i]][kdA[i]] = hi; Alo[buf][mA[i]][kdA[i]] = lo;
        }
        #pragma unroll
        for (int i = 0; i < 8; ++i) {
            unsigned int lo, hi = splitPack(bS[i][0], bS[i][1], lo);
            Bhi[buf][nB[i]][kdB[i]] = hi; Blo[buf][nB[i]][kdB[i]] = lo;
        }
        __syncthreads();
        if (kt + 1 < NKT) ISSUE_FB();

        int arow0 = waveM * 32 + l16, arow1 = arow0 + 16;
        int aoff = half * 4;
        FragU a0h, a0l, a1h, a1l;
        *(uint4*)&a0h.u[0] = *(const uint4*)&Ahi[buf][arow0][aoff];
        *(uint4*)&a0h.u[4] = *(const uint4*)&Ahi[buf][arow0][aoff + 8];
        *(uint4*)&a0l.u[0] = *(const uint4*)&Alo[buf][arow0][aoff];
        *(uint4*)&a0l.u[4] = *(const uint4*)&Alo[buf][arow0][aoff + 8];
        *(uint4*)&a1h.u[0] = *(const uint4*)&Ahi[buf][arow1][aoff];
        *(uint4*)&a1h.u[4] = *(const uint4*)&Ahi[buf][arow1][aoff + 8];
        *(uint4*)&a1l.u[0] = *(const uint4*)&Alo[buf][arow1][aoff];
        *(uint4*)&a1l.u[4] = *(const uint4*)&Alo[buf][arow1][aoff + 8];
        int brow0 = waveN * 32 + l16, brow1 = brow0 + 16;
        int boff = half * 4;
        FragU b0h, b0l, b1h, b1l;
        *(uint4*)&b0h.u[0] = *(const uint4*)&Bhi[buf][brow0][boff];
        *(uint4*)&b0h.u[4] = *(const uint4*)&Bhi[buf][brow0][boff + 8];
        *(uint4*)&b0l.u[0] = *(const uint4*)&Blo[buf][brow0][boff];
        *(uint4*)&b0l.u[4] = *(const uint4*)&Blo[buf][brow0][boff + 8];
        *(uint4*)&b1h.u[0] = *(const uint4*)&Bhi[buf][brow1][boff];
        *(uint4*)&b1h.u[4] = *(const uint4*)&Bhi[buf][brow1][boff + 8];
        *(uint4*)&b1l.u[0] = *(const uint4*)&Blo[buf][brow1][boff];
        *(uint4*)&b1l.u[4] = *(const uint4*)&Blo[buf][brow1][boff + 8];

        acc00 = __builtin_amdgcn_wmma_f32_16x16x32_f16(false, a0h.v, false, b0h.v, (short)0, acc00, false, false);
        acc00 = __builtin_amdgcn_wmma_f32_16x16x32_f16(false, a0h.v, false, b0l.v, (short)0, acc00, false, false);
        acc00 = __builtin_amdgcn_wmma_f32_16x16x32_f16(false, a0l.v, false, b0h.v, (short)0, acc00, false, false);
        acc01 = __builtin_amdgcn_wmma_f32_16x16x32_f16(false, a0h.v, false, b1h.v, (short)0, acc01, false, false);
        acc01 = __builtin_amdgcn_wmma_f32_16x16x32_f16(false, a0h.v, false, b1l.v, (short)0, acc01, false, false);
        acc01 = __builtin_amdgcn_wmma_f32_16x16x32_f16(false, a0l.v, false, b1h.v, (short)0, acc01, false, false);
        acc10 = __builtin_amdgcn_wmma_f32_16x16x32_f16(false, a1h.v, false, b0h.v, (short)0, acc10, false, false);
        acc10 = __builtin_amdgcn_wmma_f32_16x16x32_f16(false, a1h.v, false, b0l.v, (short)0, acc10, false, false);
        acc10 = __builtin_amdgcn_wmma_f32_16x16x32_f16(false, a1l.v, false, b0h.v, (short)0, acc10, false, false);
        acc11 = __builtin_amdgcn_wmma_f32_16x16x32_f16(false, a1h.v, false, b1h.v, (short)0, acc11, false, false);
        acc11 = __builtin_amdgcn_wmma_f32_16x16x32_f16(false, a1h.v, false, b1l.v, (short)0, acc11, false, false);
        acc11 = __builtin_amdgcn_wmma_f32_16x16x32_f16(false, a1l.v, false, b1h.v, (short)0, acc11, false, false);
    }
    int n0 = Nbase + waveN * 32 + l16, n1 = n0 + 16;
    #pragma unroll
    for (int r = 0; r < 8; ++r) {
        int o0 = Mbase + waveM * 32 + half * 8 + r;
        int o1 = o0 + 16;
        float s0 = se[b * OCH + o0], s1 = se[b * OCH + o1];
        float bb0 = bias[o0], bb1 = bias[o1];
        size_t r0 = (size_t)(b * OCH + o0) * HWSZ;
        size_t r1 = (size_t)(b * OCH + o1) * HWSZ;
        out[r0 + n0] = (acc00[r] + bb0) * s0;
        out[r0 + n1] = (acc01[r] + bb0) * s0;
        out[r1 + n0] = (acc10[r] + bb1) * s1;
        out[r1 + n1] = (acc11[r] + bb1) * s1;
    }
    #undef ISSUE_FB
}

extern "C" void kernel_launch(void* const* d_in, const int* in_sizes, int n_in,
                              void* d_out, int out_size, void* d_ws, size_t ws_size,
                              hipStream_t stream) {
    const float* x      = (const float*)d_in[0];
    const float* weight = (const float*)d_in[1];   // expert 0 at offset 0
    const float* bias_e = (const float*)d_in[2];
    const float* se_w1  = (const float*)d_in[5];
    const float* se_w2  = (const float*)d_in[6];
    float* out = (float*)d_out;

    float*        avg = (float*)d_ws;                       // 4096 f32
    float*        se  = avg + BSZ * CCH;                    // 4096 f32
    unsigned int* Ahp = (unsigned int*)(se + BSZ * OCH);    // 128*576 dwords
    unsigned int* Alp = Ahp + OCH * KDWT;                   // 128*576 dwords
    unsigned int* xsp = Alp + OCH * KDWT;                   // B*C*H*W dwords
    size_t need = (size_t)((char*)(xsp + (size_t)BSZ * CCH * HWSZ) - (char*)d_ws);

    if (ws_size >= need) {
        avgsplit_kernel<<<BSZ * CCH, 256, 0, stream>>>(x, avg, xsp);
        aprep_kernel<<<(OCH * KDWT) / 256, 256, 0, stream>>>(weight, Ahp, Alp);
        se_kernel<<<BSZ, 128, 0, stream>>>(avg, se_w1, se_w2, se);
        conv_wmma_kernel<<<BSZ * NT2, 256, 0, stream>>>(xsp, Ahp, Alp, bias_e, se, out);
    } else {
        avg_kernel<<<BSZ * CCH, 256, 0, stream>>>(x, avg);
        se_kernel<<<BSZ, 128, 0, stream>>>(avg, se_w1, se_w2, se);
        conv_wmma_fb<<<BSZ * 2 * NT2, 256, 0, stream>>>(x, weight, bias_e, se, out);
    }
}
